// MTGRU_33638183863193
// MI455X (gfx1250) — compile-verified, matrix operand order checked by reading, not run on hardware
//
#include <hip/hip_runtime.h>
#include <math.h>

// ---------------------------------------------------------------------------
// MTGRU hypernetwork forward for MI455X (gfx1250, wave32, WMMA 16x16x32 f16).
//
// Roofline: keeping psi's Whh_b slice in f16 (134 MB) makes it resident in
// the 192 MB global L2, so the 64 decoder steps stream ~8.6 GB from L2
// instead of 17 GB from HBM. HBM traffic is ~0.3 GB total (W3 read + psi
// write). All dense GEMMs (encoder/decoder recurrences, readout) run on
// v_wmma_f32_16x16x32_f16 with fp32 accumulators.  The L2-streaming
// per-sample matvec (k_dec_hW) uses b64 f16 loads (4 k-columns per thread)
// with the broadcast h vector staged in LDS.
// ---------------------------------------------------------------------------

typedef __attribute__((ext_vector_type(16))) _Float16 v16h;
typedef __attribute__((ext_vector_type(4)))  _Float16 v4h;
typedef __attribute__((ext_vector_type(8)))  float    v8f;

static constexpr int BATCH = 64;
static constexpr int T     = 64;
static constexpr int HUMAN = 64;
static constexpr int IN    = 8;
static constexpr int HS    = 1024;
static constexpr int ES    = 1024;
static constexpr int KLAT  = 7;
static constexpr int PSI_H = 1024;
static constexpr int PSI_LR = 30;

// psi region offsets (element index within a sample's psi vector)
static constexpr int O_BH  = 1048576;                 // after 1024*1024 Whh
static constexpr int O_WIH = 1049600;                 // +1024
static constexpr int O_C   = 1057792;                 // +8192
static constexpr int O_D   = 1123328;                 // +65536
static constexpr int NPSI  = 1123776;                 // +448

// ---------------------------------------------------------------------------
// WMMA helpers.  CDNA5 16-bit A-matrix 16x32 layout (ISA 7.12.2):
//   lane L: row = L&15, half = L>>4
//   vector element e: g = e>>3, j = e&7  ->  k = g*16 + half*8 + j
// So elements 0..7 are 8 contiguous halves at k0+half*8, elements 8..15 are
// 8 contiguous halves at k0+16+half*8  => two 16-byte loads per lane.
// B (KxN) is loaded from a transposed [N,K] row-major copy so it has the
// identical per-lane pattern with "row" = output column n.
// C/D f32 (v8f): element v at lane L -> m = v + (L>>4)*8, n = L&15.
// ---------------------------------------------------------------------------
__device__ inline v16h load_frag(const _Float16* __restrict__ base, int ld,
                                 int r0, int k0) {
  const int lane = threadIdx.x & 31;
  const int r = lane & 15;
  const int half = lane >> 4;
  const _Float16* p = base + (size_t)(r0 + r) * ld + k0 + half * 8;
  v16h f;
#pragma unroll
  for (int j = 0; j < 8; ++j) f[j] = p[j];
#pragma unroll
  for (int j = 0; j < 8; ++j) f[8 + j] = p[16 + j];
  return f;
}

__device__ inline v8f wmma16(v16h a, v16h b, v8f c) {
  // (neg_a, A, neg_b, B, c_mod, C, reuse_a, reuse_b)
  return __builtin_amdgcn_wmma_f32_16x16x32_f16(false, a, false, b,
                                                (short)0, c, false, false);
}

__device__ inline float sigmoidf_(float x) { return 1.0f / (1.0f + expf(-x)); }

// ---------------------------------------------------------------------------
// Utility kernels
// ---------------------------------------------------------------------------
__global__ void k_f32_to_f16(const float* __restrict__ src,
                             _Float16* __restrict__ dst, int n) {
  int i = blockIdx.x * 256 + threadIdx.x;
  if (i < n) dst[i] = (_Float16)src[i];
}

// dst[c*rows + r] = src[r*cols + c]   (make [N,K] row-major copy of [K,N])
__global__ void k_transpose_f16(const float* __restrict__ src,
                                _Float16* __restrict__ dst, int rows, int cols) {
  int i = blockIdx.x * 256 + threadIdx.x;
  if (i < rows * cols) {
    int r = i / cols, c = i % cols;
    dst[(size_t)c * rows + r] = (_Float16)src[i];
  }
}

__global__ void k_zero_f32(float* p, int n) {
  int i = blockIdx.x * 256 + threadIdx.x;
  if (i < n) p[i] = 0.0f;
}
__global__ void k_zero_f16(_Float16* p, int n) {
  int i = blockIdx.x * 256 + threadIdx.x;
  if (i < n) p[i] = (_Float16)0.0f;
}

// ---------------------------------------------------------------------------
// Encoder GRU step (one launch per t, ping-pong h buffers).
// grid = 4 M-tiles x 64 N-tiles = 256 single-wave blocks.
// Accumulators: r,z combined over x&h (K=64 + K=1024); n gate needs
// xn and hn separately (n = tanh(xn + r*hn)).
// ---------------------------------------------------------------------------
__global__ __launch_bounds__(32)
void k_enc_step(const _Float16* __restrict__ out16, int t,
                const float* __restrict__ h_in,
                const _Float16* __restrict__ h16_in,
                const _Float16* __restrict__ Wih16,   // [3*ES, 64]
                const _Float16* __restrict__ Whh16,   // [3*ES, ES]
                const float* __restrict__ bih,
                const float* __restrict__ bhh,
                float* __restrict__ h_out,
                _Float16* __restrict__ h16_out) {
  const int tile = blockIdx.x;
  const int m0 = (tile & 3) * 16;        // batch rows
  const int n0 = (tile >> 2) * 16;       // hidden columns (within ES)

  v8f ar = {}, az = {}, axn = {}, ahn = {};
  // h part, K = 1024
  for (int k0 = 0; k0 < ES; k0 += 32) {
    v16h a  = load_frag(h16_in, ES, m0, k0);
    v16h br = load_frag(Whh16, ES, n0,            k0);
    v16h bz = load_frag(Whh16, ES, n0 + ES,       k0);
    v16h bn = load_frag(Whh16, ES, n0 + 2 * ES,   k0);
    ar  = wmma16(a, br, ar);
    az  = wmma16(a, bz, az);
    ahn = wmma16(a, bn, ahn);
  }
  // x part, K = 64 (out16 layout [B,T,HUMAN]; row stride T*HUMAN)
  for (int k0 = 0; k0 < HUMAN; k0 += 32) {
    v16h a  = load_frag(out16 + t * HUMAN, T * HUMAN, m0, k0);
    v16h br = load_frag(Wih16, HUMAN, n0,          k0);
    v16h bz = load_frag(Wih16, HUMAN, n0 + ES,     k0);
    v16h bn = load_frag(Wih16, HUMAN, n0 + 2 * ES, k0);
    ar  = wmma16(a, br, ar);
    az  = wmma16(a, bz, az);
    axn = wmma16(a, bn, axn);
  }
  const int lane = threadIdx.x & 31;
  const int n = n0 + (lane & 15);
  const int half = lane >> 4;
#pragma unroll
  for (int v = 0; v < 8; ++v) {
    const int b = m0 + v + half * 8;
    float r  = sigmoidf_(ar[v] + bih[n] + bhh[n]);
    float z  = sigmoidf_(az[v] + bih[n + ES] + bhh[n + ES]);
    float nn = tanhf(axn[v] + bih[n + 2 * ES] + r * (ahn[v] + bhh[n + 2 * ES]));
    float hold = h_in[b * ES + n];
    float hnew = (1.0f - z) * nn + z * hold;
    h_out[b * ES + n]   = hnew;
    h16_out[b * ES + n] = (_Float16)hnew;
  }
}

// ---------------------------------------------------------------------------
// Latent + hypernet head: mu/logstd/z, h1 = tanh(z@W1+b1), h2 = h1@W2+b2.
// Everything tiny; one 256-thread workgroup.
// ---------------------------------------------------------------------------
__global__ __launch_bounds__(256)
void k_latent(const float* __restrict__ h_enc, const float* __restrict__ eps,
              const float* __restrict__ to_mu, const float* __restrict__ to_ls,
              const float* __restrict__ ls_bias,
              const float* __restrict__ W1, const float* __restrict__ b1,
              const float* __restrict__ W2, const float* __restrict__ b2,
              float* __restrict__ mu_out, float* __restrict__ ls_out,
              float* __restrict__ h1_ws, float* __restrict__ h2_ws) {
  __shared__ float zsh[BATCH * KLAT];
  for (int o = threadIdx.x; o < BATCH * KLAT; o += 256) {
    int b = o / KLAT, k = o % KLAT;
    const float* h = h_enc + b * ES;
    float m = 0.0f, s = 0.0f;
    for (int j = 0; j < ES; ++j) {
      float hv = h[j];
      m += hv * to_mu[j * KLAT + k];
      s += hv * to_ls[j * KLAT + k];
    }
    s += ls_bias[k];
    mu_out[o] = m;
    ls_out[o] = s;
    zsh[o] = m + eps[o] * expf(s);
  }
  __syncthreads();
  for (int o = threadIdx.x; o < BATCH * PSI_H; o += 256) {
    int b = o >> 10, j = o & (PSI_H - 1);
    float a = b1[j];
#pragma unroll
    for (int k = 0; k < KLAT; ++k) a += zsh[b * KLAT + k] * W1[k * PSI_H + j];
    h1_ws[o] = tanhf(a);
  }
  __threadfence();
  __syncthreads();
  for (int o = threadIdx.x; o < BATCH * PSI_LR; o += 256) {
    int b = o / PSI_LR, r = o % PSI_LR;
    float a = b2[r];
    const float* h1 = h1_ws + b * PSI_H;
    for (int j = 0; j < PSI_H; ++j) a += h1[j] * W2[j * PSI_LR + r];
    h2_ws[o] = a;
  }
}

// ---------------------------------------------------------------------------
// psi = h2 @ W3 + b3, scattered into typed regions.  One thread per psi
// column n; W3 column (30 values) read once, reused for all 64 samples.
// Whh_b and C_b stored f16 (L2-resident); C_b stored transposed [HUMAN, HS].
// ---------------------------------------------------------------------------
__global__ __launch_bounds__(256)
void k_psi(const float* __restrict__ h2, const float* __restrict__ W3,
           const float* __restrict__ b3,
           _Float16* __restrict__ WhhB,   // [B][HS][HS] f16
           float* __restrict__ bhB,       // [B][HS]
           float* __restrict__ WihB,      // [B][IN][HS]
           _Float16* __restrict__ CBt,    // [B][HUMAN][HS] f16 (transposed)
           float* __restrict__ DB) {      // [B][IN][HUMAN-IN]
  __shared__ float h2s[BATCH * PSI_LR];
  for (int i = threadIdx.x; i < BATCH * PSI_LR; i += 256) h2s[i] = h2[i];
  __syncthreads();
  const int n = blockIdx.x * 256 + threadIdx.x;
  if (n >= NPSI) return;
  float w[PSI_LR];
#pragma unroll
  for (int r = 0; r < PSI_LR; ++r) w[r] = W3[(size_t)r * NPSI + n];
  const float bias = b3[n];
  for (int b = 0; b < BATCH; ++b) {
    float acc = bias;
#pragma unroll
    for (int r = 0; r < PSI_LR; ++r) acc += h2s[b * PSI_LR + r] * w[r];
    if (n < O_BH) {
      WhhB[((size_t)b << 20) + n] = (_Float16)acc;
    } else if (n < O_WIH) {
      bhB[b * HS + (n - O_BH)] = acc;
    } else if (n < O_C) {
      WihB[b * (IN * HS) + (n - O_WIH)] = acc;
    } else if (n < O_D) {
      int idx = n - O_C;
      int hh = idx >> 6, m = idx & 63;            // psi layout [HS, HUMAN]
      CBt[((size_t)b << 16) + m * HS + hh] = (_Float16)acc;
    } else {
      DB[b * (IN * (HUMAN - IN)) + (n - O_D)] = acc;
    }
  }
}

// ---------------------------------------------------------------------------
// Decoder per-sample matvec: hWtanh[b][k] = tanh(h[b]·WhhB[b][:,k] + bh[b][k])
// One block per sample; each thread owns 4 consecutive k columns and streams
// WhhB with b64 (4x f16) loads -> 256B per wave per iteration from L2.  The
// broadcast h vector is staged once in LDS.
// ---------------------------------------------------------------------------
__global__ __launch_bounds__(256)
void k_dec_hW(const float* __restrict__ h_in,
              const _Float16* __restrict__ WhhB,
              const float* __restrict__ bhB,
              float* __restrict__ hWtanh) {
  const int b  = blockIdx.x;
  const int k0 = threadIdx.x * 4;
  __shared__ float hs[HS];
  const float* h = h_in + b * HS;
  for (int i = threadIdx.x; i < HS; i += 256) hs[i] = h[i];
  __syncthreads();

  const _Float16* W = WhhB + ((size_t)b << 20) + k0;
  float a0 = 0.0f, a1 = 0.0f, a2 = 0.0f, a3 = 0.0f;
#pragma unroll 4
  for (int hh = 0; hh < HS; ++hh) {
    __builtin_prefetch(W + ((size_t)(hh + 32) << 10), 0, 3);
    v4h w = *(const v4h*)(W + ((size_t)hh << 10));
    float hv = hs[hh];
    a0 += hv * (float)w[0];
    a1 += hv * (float)w[1];
    a2 += hv * (float)w[2];
    a3 += hv * (float)w[3];
  }
  const int base = b * HS + k0;
  hWtanh[base + 0] = tanhf(a0 + bhB[base + 0]);
  hWtanh[base + 1] = tanhf(a1 + bhB[base + 1]);
  hWtanh[base + 2] = tanhf(a2 + bhB[base + 2]);
  hWtanh[base + 3] = tanhf(a3 + bhB[base + 3]);
}

// ---------------------------------------------------------------------------
// Decoder GRU step: z/r pre-activations via WMMA (B from transposed gru_Whh),
// gx_t and xW_t fused as K=8 VALU dots, then cell update + dec_seq store.
// grid = 4 M-tiles x 64 N-tiles = 256 single-wave blocks.
// ---------------------------------------------------------------------------
__global__ __launch_bounds__(32)
void k_dec_step(const float* __restrict__ inputs, int t,
                const float* __restrict__ h_in,
                const _Float16* __restrict__ h16_in,
                const _Float16* __restrict__ WhhT16,  // [2*HS, HS] (transposed)
                const float* __restrict__ gru_Wih,    // [IN, 2*HS]
                const float* __restrict__ gru_bias,   // [2*HS]
                const float* __restrict__ hWtanh,     // [B][HS]
                const float* __restrict__ WihB,       // [B][IN][HS]
                float* __restrict__ h_out,
                _Float16* __restrict__ h16_out,
                _Float16* __restrict__ dec_seq16,
                float* __restrict__ states_or_null) {
  const int tile = blockIdx.x;
  const int m0 = (tile & 3) * 16;
  const int n0 = (tile >> 2) * 16;

  v8f accz = {}, accr = {};
  for (int k0 = 0; k0 < HS; k0 += 32) {
    v16h a  = load_frag(h16_in, HS, m0, k0);
    v16h bz = load_frag(WhhT16, HS, n0,      k0);
    v16h br = load_frag(WhhT16, HS, n0 + HS, k0);
    accz = wmma16(a, bz, accz);
    accr = wmma16(a, br, accr);
  }
  const int lane = threadIdx.x & 31;
  const int n = n0 + (lane & 15);
  const int half = lane >> 4;
#pragma unroll
  for (int v = 0; v < 8; ++v) {
    const int b = m0 + v + half * 8;
    const float* x = inputs + (b * T + t) * IN;
    float gxz = 0.0f, gxr = 0.0f, xw = 0.0f;
#pragma unroll
    for (int i = 0; i < IN; ++i) {
      float xi = x[i];
      gxz += xi * gru_Wih[i * (2 * HS) + n];
      gxr += xi * gru_Wih[i * (2 * HS) + n + HS];
      xw  += xi * WihB[b * (IN * HS) + i * HS + n];
    }
    float zt = sigmoidf_(accz[v] + gxz + gru_bias[n]);
    float rt = sigmoidf_(accr[v] + gxr + gru_bias[n + HS]);
    float eta = tanhf(xw + rt * hWtanh[b * HS + n]);
    float hold = h_in[b * HS + n];
    float hnew = zt * hold + (1.0f - zt) * eta;
    h_out[b * HS + n]   = hnew;
    h16_out[b * HS + n] = (_Float16)hnew;
    dec_seq16[(b * T + t) * HS + n] = (_Float16)hnew;
    if (states_or_null) states_or_null[b * HS + n] = hnew;
  }
}

// ---------------------------------------------------------------------------
// Readout: yhats[b,t,m] = dec_seq[b]@C_b[b] + [inputs | x@D_b]  (WMMA per
// sample: M=T=64, N=HUMAN=64, K=1024).  grid = 64 samples * 16 tiles.
// ---------------------------------------------------------------------------
__global__ __launch_bounds__(32)
void k_readout(const _Float16* __restrict__ dec_seq16,
               const _Float16* __restrict__ CBt,
               const float* __restrict__ inputs,
               const float* __restrict__ DB,
               float* __restrict__ yhats) {
  const int id = blockIdx.x;
  const int b = id >> 4;
  const int tile = id & 15;
  const int m0 = (tile & 3) * 16;   // t rows
  const int n0 = (tile >> 2) * 16;  // human cols

  v8f acc = {};
  const _Float16* A = dec_seq16 + (size_t)b * T * HS;
  const _Float16* Bm = CBt + ((size_t)b << 16);
  for (int k0 = 0; k0 < HS; k0 += 32) {
    v16h a  = load_frag(A,  HS, m0, k0);
    v16h bb = load_frag(Bm, HS, n0, k0);
    acc = wmma16(a, bb, acc);
  }
  const int lane = threadIdx.x & 31;
  const int n = n0 + (lane & 15);
  const int half = lane >> 4;
#pragma unroll
  for (int v = 0; v < 8; ++v) {
    const int tt = m0 + v + half * 8;
    const float* x = inputs + (b * T + tt) * IN;
    float add;
    if (n < IN) {
      add = x[n];  // residual input
    } else {
      float s = 0.0f;
#pragma unroll
      for (int i = 0; i < IN; ++i)
        s += x[i] * DB[b * (IN * (HUMAN - IN)) + i * (HUMAN - IN) + (n - IN)];
      add = s;
    }
    yhats[(b * T + tt) * HUMAN + n] = acc[v] + add;
  }
}

// ---------------------------------------------------------------------------
// Host launcher
// ---------------------------------------------------------------------------
static inline size_t alignup(size_t x) { return (x + 255) & ~(size_t)255; }

extern "C" void kernel_launch(void* const* d_in, const int* in_sizes, int n_in,
                              void* d_out, int out_size, void* d_ws, size_t ws_size,
                              hipStream_t stream) {
  const float* inputs   = (const float*)d_in[0];   // [64,64,8]
  const float* outputs  = (const float*)d_in[1];   // [64,64,64]
  const float* eps      = (const float*)d_in[2];   // [64,7]
  const float* enc_Wih  = (const float*)d_in[3];   // [3072,64]
  const float* enc_Whh  = (const float*)d_in[4];   // [3072,1024]
  const float* enc_bih  = (const float*)d_in[5];   // [3072]
  const float* enc_bhh  = (const float*)d_in[6];   // [3072]
  const float* to_mu    = (const float*)d_in[7];   // [1024,7]
  const float* to_ls    = (const float*)d_in[8];   // [1024,7]
  const float* ls_bias  = (const float*)d_in[9];   // [7]
  const float* W1       = (const float*)d_in[10];  // [7,1024]
  const float* b1       = (const float*)d_in[11];  // [1024]
  const float* W2       = (const float*)d_in[12];  // [1024,30]
  const float* b2       = (const float*)d_in[13];  // [30]
  const float* W3       = (const float*)d_in[14];  // [30, NPSI]
  const float* b3       = (const float*)d_in[15];  // [NPSI]
  const float* gru_Wih  = (const float*)d_in[16];  // [8,2048]
  const float* gru_Whh  = (const float*)d_in[17];  // [1024,2048]
  const float* gru_bias = (const float*)d_in[18];  // [2048]
  (void)in_sizes; (void)n_in; (void)out_size; (void)ws_size;

  float* out = (float*)d_out;
  float* yhats  = out;                       // 64*64*64 = 262144
  float* mu_o   = out + 262144;              // 448
  float* ls_o   = out + 262592;              // 448
  float* states = out + 263040;              // 65536

  // workspace carve-up
  char* ws = (char*)d_ws;
  size_t off = 0;
  auto take = [&](size_t bytes) { char* p = ws + off; off += alignup(bytes); return p; };

  _Float16* out16   = (_Float16*)take((size_t)BATCH * T * HUMAN * 2);
  _Float16* Wih16   = (_Float16*)take((size_t)3 * ES * HUMAN * 2);
  _Float16* Whh16   = (_Float16*)take((size_t)3 * ES * ES * 2);
  _Float16* WhhT16  = (_Float16*)take((size_t)2 * HS * HS * 2);
  float*    hA      = (float*)take((size_t)BATCH * ES * 4);
  float*    hB      = (float*)take((size_t)BATCH * ES * 4);
  _Float16* h16A    = (_Float16*)take((size_t)BATCH * ES * 2);
  _Float16* h16B    = (_Float16*)take((size_t)BATCH * ES * 2);
  float*    hdA     = (float*)take((size_t)BATCH * HS * 4);
  float*    hdB     = (float*)take((size_t)BATCH * HS * 4);
  _Float16* hd16A   = (_Float16*)take((size_t)BATCH * HS * 2);
  _Float16* hd16B   = (_Float16*)take((size_t)BATCH * HS * 2);
  float*    h1_ws   = (float*)take((size_t)BATCH * PSI_H * 4);
  float*    h2_ws   = (float*)take((size_t)BATCH * PSI_LR * 4);
  _Float16* WhhB    = (_Float16*)take((size_t)BATCH * HS * HS * 2);   // 134 MB, L2-resident
  float*    bhB     = (float*)take((size_t)BATCH * HS * 4);
  float*    WihB    = (float*)take((size_t)BATCH * IN * HS * 4);
  _Float16* CBt     = (_Float16*)take((size_t)BATCH * HUMAN * HS * 2);
  float*    DB      = (float*)take((size_t)BATCH * IN * (HUMAN - IN) * 4);
  float*    hWtanh  = (float*)take((size_t)BATCH * HS * 4);
  _Float16* dec16   = (_Float16*)take((size_t)BATCH * T * HS * 2);

  auto blocks = [](int n) { return (n + 255) / 256; };

  // ---- one-time f16 conversions ----
  k_f32_to_f16<<<blocks(BATCH * T * HUMAN), 256, 0, stream>>>(outputs, out16, BATCH * T * HUMAN);
  k_f32_to_f16<<<blocks(3 * ES * HUMAN), 256, 0, stream>>>(enc_Wih, Wih16, 3 * ES * HUMAN);
  k_f32_to_f16<<<blocks(3 * ES * ES), 256, 0, stream>>>(enc_Whh, Whh16, 3 * ES * ES);
  k_transpose_f16<<<blocks(HS * 2 * HS), 256, 0, stream>>>(gru_Whh, WhhT16, HS, 2 * HS);

  // ---- encoder GRU over T steps (ping-pong A<->B, h0 = 0 in A) ----
  k_zero_f32<<<blocks(BATCH * ES), 256, 0, stream>>>(hA, BATCH * ES);
  k_zero_f16<<<blocks(BATCH * ES), 256, 0, stream>>>(h16A, BATCH * ES);
  for (int t = 0; t < T; ++t) {
    const bool odd = (t & 1);
    k_enc_step<<<256, 32, 0, stream>>>(out16, t,
        odd ? hB : hA, odd ? h16B : h16A,
        Wih16, Whh16, enc_bih, enc_bhh,
        odd ? hA : hB, odd ? h16A : h16B);
  }
  const float* h_enc = hA;  // T=64 even -> final state back in A

  // ---- latent + hypernet head ----
  k_latent<<<1, 256, 0, stream>>>(h_enc, eps, to_mu, to_ls, ls_bias,
                                  W1, b1, W2, b2, mu_o, ls_o, h1_ws, h2_ws);

  // ---- psi materialization (f16 Whh_b / C_b, f32 rest) ----
  k_psi<<<blocks(NPSI), 256, 0, stream>>>(h2_ws, W3, b3, WhhB, bhB, WihB, CBt, DB);

  // ---- decoder GRU over T steps ----
  k_zero_f32<<<blocks(BATCH * HS), 256, 0, stream>>>(hdA, BATCH * HS);
  k_zero_f16<<<blocks(BATCH * HS), 256, 0, stream>>>(hd16A, BATCH * HS);
  for (int t = 0; t < T; ++t) {
    const bool odd = (t & 1);
    const float* hin = odd ? hdB : hdA;
    const _Float16* h16in = odd ? hd16B : hd16A;
    k_dec_hW<<<BATCH, 256, 0, stream>>>(hin, WhhB, bhB, hWtanh);
    k_dec_step<<<256, 32, 0, stream>>>(inputs, t, hin, h16in,
        WhhT16, gru_Wih, gru_bias, hWtanh, WihB,
        odd ? hdA : hdB, odd ? hd16A : hd16B, dec16,
        (t == T - 1) ? states : nullptr);
  }

  // ---- readout ----
  k_readout<<<BATCH * 16, 32, 0, stream>>>(dec16, CBt, inputs, DB, yhats);
}
